// GlobalPointer_34144990003257
// MI455X (gfx1250) — matrix operand order repaired and checked by reading, not compiled, not run on hardware
//
#include <hip/hip_runtime.h>

typedef __attribute__((ext_vector_type(16))) __bf16 v16bf;
typedef __attribute__((ext_vector_type(8)))  __bf16 v8bf;
typedef __attribute__((ext_vector_type(8)))  float  v8f;

#define ENT   8
#define DIN   64
#define HID   768
#define BB    32
#define LL    512
#define NEGC  1000000000000.0f

// Pack 8 fp32 -> 8 bf16 and store as a single 16B LDS write (ds_store_b128).
__device__ __forceinline__ void pack_store8(__bf16* dst, const float4& a,
                                            const float4& b) {
    union { v8bf v; uint4 u; } p;
    p.v[0] = (__bf16)a.x; p.v[1] = (__bf16)a.y;
    p.v[2] = (__bf16)a.z; p.v[3] = (__bf16)a.w;
    p.v[4] = (__bf16)b.x; p.v[5] = (__bf16)b.y;
    p.v[6] = (__bf16)b.z; p.v[7] = (__bf16)b.w;
    *(uint4*)dst = p.u;
}

// ---------------------------------------------------------------------------
// Kernel 1: out = hidden @ W^T + b, fused RoPE, emit Q/K as bf16 [B][H][L][64]
// Block: 256 threads (8 waves). Tile: 128(M) x 128(N), BK=32, software-
// pipelined global->reg->LDS staging (b128 stores), bf16 WMMA compute.
// ---------------------------------------------------------------------------
__launch_bounds__(256)
__global__ void gemm_rope_kernel(const float* __restrict__ hidden,
                                 const float* __restrict__ W,
                                 const float* __restrict__ bias,
                                 __bf16* __restrict__ Q,
                                 __bf16* __restrict__ Kq)
{
    __shared__ __align__(16) __bf16 Abuf[128][32];   // hidden tile (m,k) bf16
    __shared__ __align__(16) __bf16 Bbuf[128][32];   // W tile (n,k) bf16
    __shared__ __align__(16) float  Cbuf[64][132];   // epilogue staging (half)

    const int tid  = threadIdx.x;
    const int lane = tid & 31;
    const int wid  = tid >> 5;
    const int half = (lane >> 4) & 1;
    const int l16  = lane & 15;
    const int wm   = (wid & 1) * 64;   // wave M offset in block tile
    const int wn   = (wid >> 1) * 32;  // wave N offset in block tile

    const int m0   = blockIdx.y * 128;   // global row (b*L + s)
    const int head = blockIdx.x;         // 0..7
    const int n0   = head * 128;         // global out-channel

    // Each thread owns 2 8-element segments per tile (128 rows x 4 segs).
    const int e0 = tid, e1 = tid + 256;
    const int r0 = e0 >> 2, c0s = (e0 & 3) << 3;
    const int r1 = e1 >> 2, c1s = (e1 & 3) << 3;

    float4 aR[2][2], bR[2][2];
    auto load_tiles = [&](int kt) {
        const float* ap0 = &hidden[(long)(m0 + r0) * HID + kt + c0s];
        const float* ap1 = &hidden[(long)(m0 + r1) * HID + kt + c1s];
        const float* bp0 = &W[(long)(n0 + r0) * HID + kt + c0s];
        const float* bp1 = &W[(long)(n0 + r1) * HID + kt + c1s];
        aR[0][0] = *(const float4*)(ap0); aR[0][1] = *(const float4*)(ap0 + 4);
        aR[1][0] = *(const float4*)(ap1); aR[1][1] = *(const float4*)(ap1 + 4);
        bR[0][0] = *(const float4*)(bp0); bR[0][1] = *(const float4*)(bp0 + 4);
        bR[1][0] = *(const float4*)(bp1); bR[1][1] = *(const float4*)(bp1 + 4);
    };

    v8f acc[4][2] = {};
    union Frag { v16bf v; unsigned int u[8]; };

    load_tiles(0);                       // prologue: tile 0 in flight

    for (int kt = 0; kt < HID; kt += 32) {
        // ---- drain staged registers into LDS (4x ds_store_b128/thread) --
        pack_store8(&Abuf[r0][c0s], aR[0][0], aR[0][1]);
        pack_store8(&Abuf[r1][c1s], aR[1][0], aR[1][1]);
        pack_store8(&Bbuf[r0][c0s], bR[0][0], bR[0][1]);
        pack_store8(&Bbuf[r1][c1s], bR[1][0], bR[1][1]);
        __syncthreads();

        // ---- prefetch next tile while this one computes ------------------
        if (kt + 32 < HID) load_tiles(kt + 32);

        // ---- A fragments: 16x32 bf16, per documented lane layout ---------
        v16bf afrag[4];
        #pragma unroll
        for (int mi = 0; mi < 4; ++mi) {
            Frag f;
            const unsigned int* row =
                (const unsigned int*)(&Abuf[wm + mi * 16 + l16][0]);
            #pragma unroll
            for (int jj = 0; jj < 8; ++jj) {
                int k = ((jj >> 2) << 4) + half * 8 + ((jj & 3) << 1);
                f.u[jj] = row[k >> 1];
            }
            afrag[mi] = f.v;
        }
        // ---- B fragments: lane holds n=l16, K in same order as A ---------
        v16bf bfrag[2];
        #pragma unroll
        for (int ni = 0; ni < 2; ++ni) {
            Frag f;
            const unsigned int* row =
                (const unsigned int*)(&Bbuf[wn + ni * 16 + l16][0]);
            #pragma unroll
            for (int jj = 0; jj < 8; ++jj) f.u[jj] = row[((jj >> 2) << 3) + half * 4 + (jj & 3)];
            bfrag[ni] = f.v;
        }
        #pragma unroll
        for (int mi = 0; mi < 4; ++mi)
            #pragma unroll
            for (int ni = 0; ni < 2; ++ni)
                acc[mi][ni] = __builtin_amdgcn_wmma_f32_16x16x32_bf16(
                    false, afrag[mi], false, bfrag[ni],
                    (short)0, acc[mi][ni], false, false);
        __syncthreads();
    }

    // ---- epilogue: bias + RoPE, in two 64-row halves through LDS --------
    const float LOG1E4_OVER_32 = 0.2878231366242556f; // ln(10000)/32
    for (int pass = 0; pass < 2; ++pass) {
        __syncthreads();
        if ((wid & 1) == pass) {
            #pragma unroll
            for (int mi = 0; mi < 4; ++mi)
                #pragma unroll
                for (int ni = 0; ni < 2; ++ni)
                    #pragma unroll
                    for (int v = 0; v < 8; ++v)
                        Cbuf[mi * 16 + half * 8 + v][wn + ni * 16 + l16] =
                            acc[mi][ni][v];
        }
        __syncthreads();
        // 64 rows * 64 channel-pairs = 4096 pairs; 16 per thread
        #pragma unroll 4
        for (int i = 0; i < 16; ++i) {
            int p  = tid + i * 256;
            int r  = p >> 6;
            int cc = (p & 63) << 1;
            int m  = m0 + pass * 64 + r;
            int bi = m / LL;
            int s  = m - bi * LL;
            float x0 = Cbuf[r][cc]     + bias[n0 + cc];
            float x1 = Cbuf[r][cc + 1] + bias[n0 + cc + 1];
            int cq = cc & 63;                 // even d within q or k half
            int j  = cq >> 1;
            float freq = __expf(-(float)j * LOG1E4_OVER_32); // 10000^(-2j/64)
            float ang  = (float)s * freq;
            float sn, cs;
            __sincosf(ang, &sn, &cs);
            float y0 = x0 * cs - x1 * sn;
            float y1 = x1 * cs + x0 * sn;
            __bf16* dst = (cc < DIN) ? Q : Kq;
            long base = (((long)bi * ENT + head) * LL + s) * DIN + cq;
            dst[base]     = (__bf16)y0;
            dst[base + 1] = (__bf16)y1;
        }
    }
}

// ---------------------------------------------------------------------------
// Kernel 2: logits[b,h,m,n] = (mask-adjusted) <q[b,h,m,:], k[b,h,n,:]> / 8
// Block: 256 threads, tile 128x128 per (b,h); K=64 (two WMMA steps).
// Fragments loaded straight from global (Q/K are 32MB total -> live in L2).
// ---------------------------------------------------------------------------
__launch_bounds__(256)
__global__ void attn_logits_kernel(const __bf16* __restrict__ Q,
                                   const __bf16* __restrict__ Kq,
                                   const int* __restrict__ mask,
                                   float* __restrict__ out)
{
    const int tid  = threadIdx.x;
    const int lane = tid & 31;
    const int wid  = tid >> 5;
    const int half = (lane >> 4) & 1;
    const int l16  = lane & 15;
    const int wm   = (wid & 1) * 64;
    const int wn   = (wid >> 1) * 32;

    const int bh = blockIdx.z;        // b*8 + h
    const int bi = bh >> 3;
    const int m0 = blockIdx.y * 128;
    const int n0 = blockIdx.x * 128;

    const __bf16* Qb = Q  + (long)bh * LL * DIN;
    const __bf16* Kb = Kq + (long)bh * LL * DIN;

    v8f acc[4][2] = {};

    union U16 { v16bf v16; v8bf v8[2]; };

    #pragma unroll
    for (int kk = 0; kk < DIN; kk += 32) {
        v16bf afrag[4];
        #pragma unroll
        for (int mi = 0; mi < 4; ++mi) {
            int m = m0 + wm + mi * 16 + l16;
            const __bf16* rp = Qb + (long)m * DIN + kk + half * 8;
            U16 u;
            u.v8[0] = *(const v8bf*)(rp);        // K = kk+half*8    .. +7
            u.v8[1] = *(const v8bf*)(rp + 16);   // K = kk+16+half*8 .. +7
            afrag[mi] = u.v16;
        }
        v16bf bfrag[2];
        #pragma unroll
        for (int ni = 0; ni < 2; ++ni) {
            int n = n0 + wn + ni * 16 + l16;
            const __bf16* rp = Kb + (long)n * DIN + kk + half * 8;
            U16 u;
            u.v8[0] = *(const v8bf*)(rp);        // K = kk+half*8    .. +7
            u.v8[1] = *(const v8bf*)(rp + 16);   // K = kk+16+half*8 .. +7
            bfrag[ni] = u.v16;
        }
        #pragma unroll
        for (int mi = 0; mi < 4; ++mi)
            #pragma unroll
            for (int ni = 0; ni < 2; ++ni)
                acc[mi][ni] = __builtin_amdgcn_wmma_f32_16x16x32_bf16(
                    false, afrag[mi], false, bfrag[ni],
                    (short)0, acc[mi][ni], false, false);
    }

    // ---- masking epilogue (exact reference order) ------------------------
    #pragma unroll
    for (int mi = 0; mi < 4; ++mi)
        #pragma unroll
        for (int ni = 0; ni < 2; ++ni) {
            int n = n0 + wn + ni * 16 + l16;
            float pm  = (float)mask[bi * LL + n];
            float neg = (1.0f - pm) * NEGC;
            #pragma unroll
            for (int v = 0; v < 8; ++v) {
                int m = m0 + wm + mi * 16 + half * 8 + v;
                float val = acc[mi][ni][v];
                val = val * pm - neg;          // attention mask
                if (n < m) val -= NEGC;        // tril(..., -1) causal mask
                val *= 0.125f;                 // 1/sqrt(64)
                out[((long)bh * LL + m) * LL + n] = val;
            }
        }
}

extern "C" void kernel_launch(void* const* d_in, const int* in_sizes, int n_in,
                              void* d_out, int out_size, void* d_ws, size_t ws_size,
                              hipStream_t stream) {
    const float* hidden = (const float*)d_in[0];
    const int*   mask   = (const int*)d_in[1];
    const float* W      = (const float*)d_in[2];
    const float* bias   = (const float*)d_in[3];
    float* out = (float*)d_out;

    __bf16* Q  = (__bf16*)d_ws;                       // 16 MB
    __bf16* Kq = Q + (size_t)BB * ENT * LL * DIN;     // 16 MB

    dim3 g1(ENT, (BB * LL) / 128);        // (8, 128)
    gemm_rope_kernel<<<g1, 256, 0, stream>>>(hidden, W, bias, Q, Kq);

    dim3 g2(LL / 128, LL / 128, BB * ENT); // (4, 4, 256)
    attn_logits_kernel<<<g2, 256, 0, stream>>>(Q, Kq, mask, out);
}
